// MoSA_60885456388859
// MI455X (gfx1250) — matrix operand order repaired and checked by reading, not compiled, not run on hardware
//
#include <hip/hip_runtime.h>
#include <stdint.h>

// ---- problem constants ------------------------------------------------------
#define Bb    2
#define Tt    2048
#define Hh    1024
#define HPd   64
#define NHM   8
#define NHD   8
#define KSEL  256      // T / SPARSITY
#define RHALF 16       // N_ROT / 2
#define CTXW  512      // NH * HP
#define LOG2_BASE 13.287712379549449f   // log2(10000)

// ---- WMMA (gfx1250, wave32) -------------------------------------------------
typedef __attribute__((ext_vector_type(16))) __bf16 v16bf;
typedef __attribute__((ext_vector_type(8)))  float  v8f;
typedef __attribute__((ext_vector_type(4)))  int    v4i;

__device__ __forceinline__ __bf16 to_bf16(float f) { return (__bf16)f; }
__device__ __forceinline__ float bf16_to_f32(__bf16 b) { return (float)b; }

__device__ __forceinline__ v8f wmma_bf16(v16bf a, v16bf b, v8f c) {
  // D(16x16 f32) = A(16x32 bf16) * B(32x16 bf16) + C
  return __builtin_amdgcn_wmma_f32_16x16x32_bf16(false, a, false, b,
                                                 (short)0, c, false, false);
}

// A fragment: ISA 16-bit A 16x32 layout. lane%16 = M row; lane/16 selects K
// bank: VGPR v holds K = (v/4)*16 + (lane/16)*8 + (v%4)*2 + {0,1}.
template <typename F>
__device__ __forceinline__ v16bf load_a_frag(F&& elem /*(m,k)->float*/) {
  const int lane = threadIdx.x & 31;
  const int m = lane & 15, g = lane >> 4;
  v16bf a;
#pragma unroll
  for (int v = 0; v < 8; ++v) {
    const int kb = (v >> 2) * 16 + g * 8 + (v & 3) * 2;
    a[2 * v]     = to_bf16(elem(m, kb));
    a[2 * v + 1] = to_bf16(elem(m, kb + 1));
  }
  return a;
}

// B fragment: lane%16 = N column; K order same as A fragment.
template <typename F>
__device__ __forceinline__ v16bf load_b_frag(F&& elem /*(k,n)->float*/) {
  const int lane = threadIdx.x & 31;
  const int n = lane & 15, g = lane >> 4;
  v16bf b;
#pragma unroll
  for (int v = 0; v < 8; ++v) {
    const int kb = (v >> 2) * 16 + g * 8 + (v & 3) * 2;
    b[2 * v]     = to_bf16(elem(kb, n));
    b[2 * v + 1] = to_bf16(elem(kb + 1, n));
  }
  return b;
}

// C/D: lane%16 = N, VGPR r -> M = r + 8*(lane/16).
template <typename F>
__device__ __forceinline__ void store_c(const v8f& c, F&& st /*(m,n,val)*/) {
  const int lane = threadIdx.x & 31;
  const int n = lane & 15, mh = (lane >> 4) * 8;
#pragma unroll
  for (int r = 0; r < 8; ++r) st(mh + r, n, c[r]);
}

// ---- CDNA5 async global->LDS staging (ASYNCcnt path) ------------------------
__device__ __forceinline__ void async_copy_b128(unsigned lds_off, const void* g) {
  asm volatile("global_load_async_to_lds_b128 %0, %1, off"
               :: "v"(lds_off), "v"(g)
               : "memory");
}
__device__ __forceinline__ void wait_async0() {
  asm volatile("s_wait_asynccnt 0" ::: "memory");
}
__device__ __forceinline__ void wait_ds0() {
  asm volatile("s_wait_dscnt 0" ::: "memory");
}
__device__ __forceinline__ unsigned lds_addr_of(const void* p) {
  return (unsigned)(size_t)p;   // generic LDS address: low 32 bits = LDS offset
}

// LDS 16x16 bf16 row loads (plain, no transpose) -> one 32x16 WMMA B operand (2 tiles).
// Source tile row-major [row][col], ld = row stride in elements; result is the
// B fragment: n = row - row0, k = col - col0 in table order (plain ds_load_b128).
__device__ __forceinline__ v16bf load_b_frag_tr16(unsigned base, int row0,
                                                  int col0, int ld) {
  const int lane = threadIdx.x & 31;
  // each lane points at its 16-byte chunk: row = row0+lane%16,
  // col segment = col0 + (lane/16)*8  (tile0), +16 for tile1
  unsigned a0 =
      base + (unsigned)(((row0 + (lane & 15)) * ld + col0 + ((lane >> 4) * 8)) * 2);
  unsigned a1 = a0 + 32u;
  v4i lo, hi;
  asm volatile("ds_load_b128 %0, %2\n\t"
               "ds_load_b128 %1, %3\n\t"
               "s_wait_dscnt 0"
               : "=&v"(lo), "=&v"(hi)
               : "v"(a0), "v"(a1)
               : "memory");
  struct P2 { v4i a, b; } p{lo, hi};
  return __builtin_bit_cast(v16bf, p);
}

// ---- kernel 1: router scores + top-k selection (sorted by index) ------------
__global__ void k_router_topk(const float* __restrict__ X,
                              const float* __restrict__ rw,
                              int* __restrict__ idx_out,
                              float* __restrict__ gate_out) {
  const int b = blockIdx.x / NHM, n = blockIdx.x % NHM;
  __shared__ float sc[Tt];
  __shared__ unsigned char sel[Tt];
  __shared__ int pfx[Tt];
  const float* Xb = X + (size_t)b * Tt * Hh;
  for (int t = threadIdx.x; t < Tt; t += blockDim.x) {
    const float* xr = Xb + (size_t)t * Hh;
    float s = 0.f;
    for (int h = 0; h < Hh; ++h) s += xr[h] * rw[h * NHM + n];
    sc[t] = s;
  }
  __syncthreads();
  for (int t = threadIdx.x; t < Tt; t += blockDim.x) {
    const float st = sc[t];
    int rank = 0;
    for (int s2 = 0; s2 < Tt; ++s2) {
      const float v = sc[s2];
      rank += (v > st) || (v == st && s2 < t);   // top_k tie-break: first wins
    }
    sel[t] = (rank < KSEL) ? 1 : 0;
  }
  __syncthreads();
  if (threadIdx.x == 0) {
    int acc = 0;
    for (int t = 0; t < Tt; ++t) { pfx[t] = acc; acc += sel[t]; }
  }
  __syncthreads();
  for (int t = threadIdx.x; t < Tt; t += blockDim.x) {
    if (sel[t]) {
      const int p = pfx[t];
      idx_out[(b * NHM + n) * KSEL + p]  = t;
      gate_out[(b * NHM + n) * KSEL + p] = 1.f / (1.f + __expf(-sc[t]));
    }
  }
}

// ---- zero a float buffer ----------------------------------------------------
__global__ void k_zero(float* __restrict__ p, int n) {
  const int i = blockIdx.x * blockDim.x + threadIdx.x;
  for (int j = i; j < n; j += gridDim.x * blockDim.x) p[j] = 0.f;
}

// ---- kernel 2: MoSA gathered QKV projection (WMMA) --------------------------
// grid (16 mtiles, 12 ntiles, B*NHM); block 32 (one wave = one 16x16 tile)
__global__ void k_mosa_qkv(const float* __restrict__ X,
                           const float* __restrict__ wqkv,
                           const int* __restrict__ idx,
                           __bf16* __restrict__ Q, __bf16* __restrict__ K,
                           __bf16* __restrict__ V) {
  const int mt = blockIdx.x, nt = blockIdx.y;
  const int b = blockIdx.z / NHM, n = blockIdx.z % NHM;
  const int lane = threadIdx.x & 31;
  const int* id = idx + (b * NHM + n) * KSEL + mt * 16;
  const int row = id[lane & 15];
  const float* xr = X + ((size_t)b * Tt + row) * Hh;    // my A row
  v8f acc = {};
  for (int kc = 0; kc < Hh; kc += 32) {
    __builtin_prefetch(xr + kc + 128, 0, 1);
    v16bf a = load_a_frag([&](int, int kk) { return xr[kc + kk]; });
    v16bf w = load_b_frag([&](int kk, int nn) {
      return wqkv[((size_t)(kc + kk) * NHM + n) * 192 + nt * 16 + nn];
    });
    acc = wmma_bf16(a, w, acc);
  }
  __bf16* dst = (nt < 4) ? Q : (nt < 8) ? K : V;
  const int d0 = (nt & 3) * 16;
  store_c(acc, [&](int m, int nn, float val) {
    dst[(((size_t)(b * NHM + n)) * KSEL + mt * 16 + m) * HPd + d0 + nn] =
        to_bf16(val);
  });
}

// ---- kernel 3: MoSA RoPE (in place on bf16 Q/K, pos = gathered index) -------
__global__ void k_mosa_rope(__bf16* __restrict__ Q, __bf16* __restrict__ K,
                            const int* __restrict__ idx) {
  const int g = blockIdx.x * blockDim.x + threadIdx.x;
  if (g >= Bb * NHM * KSEL * RHALF) return;
  const int j = g & 15, row = g >> 4;
  const float pos = (float)idx[row];
  const float inv = exp2f(-(float)j * (LOG2_BASE / (float)RHALF));
  const float c = __cosf(pos * inv), s = __sinf(pos * inv);
  const size_t base = (size_t)row * HPd;
  {
    float x1 = bf16_to_f32(Q[base + j]), x2 = bf16_to_f32(Q[base + j + RHALF]);
    Q[base + j] = to_bf16(x1 * c - x2 * s);
    Q[base + j + RHALF] = to_bf16(x1 * s + x2 * c);
  }
  {
    float x1 = bf16_to_f32(K[base + j]), x2 = bf16_to_f32(K[base + j + RHALF]);
    K[base + j] = to_bf16(x1 * c - x2 * s);
    K[base + j + RHALF] = to_bf16(x1 * s + x2 * c);
  }
}

// ---- kernel 4: MoSA attention (256x256, sorted idx => triangular causal) ----
// grid = B*NHM*16 q-tiles; block 32 (one wave)
__global__ void k_mosa_attn(const __bf16* __restrict__ Q,
                            const __bf16* __restrict__ K,
                            const __bf16* __restrict__ V,
                            const int* __restrict__ idx,
                            const float* __restrict__ gates,
                            float* __restrict__ ctx) {
  __shared__ float S[16][KSEL];
  __shared__ float rowscale[16];
  const int qt = blockIdx.x & 15, bn = blockIdx.x >> 4;
  const int b = bn / NHM, n = bn % NHM;
  const int lane = threadIdx.x & 31;
  const __bf16* Qb = Q + (size_t)bn * KSEL * HPd;
  const __bf16* Kb = K + (size_t)bn * KSEL * HPd;
  const __bf16* Vb = V + (size_t)bn * KSEL * HPd;
  const __bf16* qr = Qb + (size_t)(qt * 16 + (lane & 15)) * HPd;
  v16bf qa[2];
  qa[0] = load_a_frag([&](int, int kk) { return bf16_to_f32(qr[kk]); });
  qa[1] = load_a_frag([&](int, int kk) { return bf16_to_f32(qr[32 + kk]); });

  for (int jt = 0; jt < KSEL / 16; ++jt) {
    v8f sa = {};
#pragma unroll
    for (int c2 = 0; c2 < 2; ++c2) {
      v16bf kb = load_b_frag([&](int kk, int nn) {
        return bf16_to_f32(Kb[(size_t)(jt * 16 + nn) * HPd + c2 * 32 + kk]);
      });
      sa = wmma_bf16(qa[c2], kb, sa);
    }
    store_c(sa, [&](int m, int nn, float val) {
      const int qp = qt * 16 + m, jp = jt * 16 + nn;
      S[m][jp] = (jp <= qp) ? val * 0.125f : -3.0e38f;
    });
  }
  __syncthreads();
  if (lane < 16) {
    const int qp = qt * 16 + lane;
    float mx = -3.0e38f;
    for (int j2 = 0; j2 < KSEL; ++j2) mx = fmaxf(mx, S[lane][j2]);
    float sum = 0.f;
    for (int j2 = 0; j2 < KSEL; ++j2) {
      const float e = __expf(S[lane][j2] - mx);
      S[lane][j2] = e;
      sum += e;
    }
    rowscale[lane] = gates[bn * KSEL + qp] / sum;
  }
  __syncthreads();

  v8f oacc[4] = {{}, {}, {}, {}};
  for (int kc = 0; kc < KSEL; kc += 32) {
    v16bf pa = load_a_frag([&](int m, int kk) { return S[m][kc + kk]; });
#pragma unroll
    for (int ntv = 0; ntv < 4; ++ntv) {
      v16bf vb = load_b_frag([&](int kk, int nn) {
        return bf16_to_f32(Vb[(size_t)(kc + kk) * HPd + ntv * 16 + nn]);
      });
      oacc[ntv] = wmma_bf16(pa, vb, oacc[ntv]);
    }
  }
  const int* id = idx + bn * KSEL + qt * 16;
#pragma unroll
  for (int ntv = 0; ntv < 4; ++ntv)
    store_c(oacc[ntv], [&](int m, int nn, float val) {
      const int t = id[m];
      ctx[((size_t)(b * Tt + t)) * CTXW + n * HPd + ntv * 16 + nn] =
          val * rowscale[m];
    });
}

// ---- kernel 5: dense QKV projection (WMMA GEMM 4096x1024 x 1024x1536) -------
// grid (256 row tiles, 96 col tiles); block 32
__global__ void k_dense_qkv(const float* __restrict__ X,
                            const float* __restrict__ wqkv,
                            __bf16* __restrict__ Q, __bf16* __restrict__ K,
                            __bf16* __restrict__ V) {
  const int rt = blockIdx.x, ct = blockIdx.y;
  const int head = ct / 12, ntl = ct % 12;
  const int lane = threadIdx.x & 31;
  const float* xr = X + (size_t)(rt * 16 + (lane & 15)) * Hh;
  v8f acc = {};
  for (int kc = 0; kc < Hh; kc += 32) {
    __builtin_prefetch(xr + kc + 128, 0, 1);
    v16bf a = load_a_frag([&](int, int kk) { return xr[kc + kk]; });
    v16bf w = load_b_frag([&](int kk, int nn) {
      return wqkv[((size_t)(kc + kk) * NHD + head) * 192 + ntl * 16 + nn];
    });
    acc = wmma_bf16(a, w, acc);
  }
  __bf16* dst = (ntl < 4) ? Q : (ntl < 8) ? K : V;
  const int d0 = (ntl & 3) * 16;
  store_c(acc, [&](int m, int nn, float val) {
    const int r = rt * 16 + m, b = r / Tt, t = r % Tt;
    dst[((size_t)(b * NHD + head) * Tt + t) * HPd + d0 + nn] = to_bf16(val);
  });
}

// ---- kernel 6: dense RoPE (pos = t) -----------------------------------------
__global__ void k_dense_rope(__bf16* __restrict__ Q, __bf16* __restrict__ K) {
  const int g = blockIdx.x * blockDim.x + threadIdx.x;
  if (g >= Bb * NHD * Tt * RHALF) return;
  const int j = g & 15, row = g >> 4;
  const float pos = (float)(row % Tt);
  const float inv = exp2f(-(float)j * (LOG2_BASE / (float)RHALF));
  const float c = __cosf(pos * inv), s = __sinf(pos * inv);
  const size_t base = (size_t)row * HPd;
  {
    float x1 = bf16_to_f32(Q[base + j]), x2 = bf16_to_f32(Q[base + j + RHALF]);
    Q[base + j] = to_bf16(x1 * c - x2 * s);
    Q[base + j + RHALF] = to_bf16(x1 * s + x2 * c);
  }
  {
    float x1 = bf16_to_f32(K[base + j]), x2 = bf16_to_f32(K[base + j + RHALF]);
    K[base + j] = to_bf16(x1 * c - x2 * s);
    K[base + j + RHALF] = to_bf16(x1 * s + x2 * c);
  }
}

// ---- kernel 7: dense causal attention, flash-style online softmax -----------
// grid (128 q-tiles, B*NHD); block 32 (one wave).
// K/V tiles (32x64 bf16 = 4KB, contiguous) staged HBM->LDS with the CDNA5
// async DMA path (global_load_async_to_lds_b128 / s_wait_asynccnt); the QK^T
// B-operands come out of LDS through plain ds_load_b128 row loads.
__global__ void k_dense_attn(const __bf16* __restrict__ Q,
                             const __bf16* __restrict__ K,
                             const __bf16* __restrict__ V,
                             float* __restrict__ ctx) {
  const int qt = blockIdx.x, bn = blockIdx.y;
  const int b = bn / NHD, n = bn % NHD;
  const int lane = threadIdx.x & 31;
  __shared__ __bf16 kbuf[32 * HPd];   // 4KB staged K tile (rows j0..j0+31)
  __shared__ __bf16 vbuf[32 * HPd];   // 4KB staged V tile
  __shared__ float S[16][32];
  __shared__ __bf16 P[16][32];
  __shared__ float f_corr[16], f_inv[16];
  const __bf16* Qb = Q + (size_t)bn * Tt * HPd;
  const __bf16* Kb = K + (size_t)bn * Tt * HPd;
  const __bf16* Vb = V + (size_t)bn * Tt * HPd;
  const unsigned kbase = lds_addr_of(kbuf);
  const unsigned vbase = lds_addr_of(vbuf);
  const __bf16* qr = Qb + (size_t)(qt * 16 + (lane & 15)) * HPd;
  v16bf qa[2];
  qa[0] = load_a_frag([&](int, int kk) { return bf16_to_f32(qr[kk]); });
  qa[1] = load_a_frag([&](int, int kk) { return bf16_to_f32(qr[32 + kk]); });
  v8f oacc[4] = {{}, {}, {}, {}};
  float run_max = -3.0e38f, run_sum = 0.f;   // lanes 0..15 own rows
  const int qhi = qt * 16 + 15;

  for (int j0 = 0; j0 <= qhi; j0 += 32) {
    // drain reads of previous tile, then kick the async DMA for this tile
    wait_ds0();
    const char* kg = (const char*)(Kb + (size_t)j0 * HPd);
    const char* vg = (const char*)(Vb + (size_t)j0 * HPd);
#pragma unroll
    for (int i = 0; i < 8; ++i) {
      const int off = i * 512 + lane * 16;
      async_copy_b128(kbase + off, kg + off);
      async_copy_b128(vbase + off, vg + off);
    }
    wait_async0();
    __syncthreads();

#pragma unroll
    for (int st = 0; st < 2; ++st) {
      v8f sa = {};
#pragma unroll
      for (int c2 = 0; c2 < 2; ++c2) {
        // transposed 32x16 B operand from the staged K tile
        v16bf kb = load_b_frag_tr16(kbase, st * 16, c2 * 32, HPd);
        sa = wmma_bf16(qa[c2], kb, sa);
      }
      store_c(sa, [&](int m, int nn, float val) {
        const int qp = qt * 16 + m, jp = j0 + st * 16 + nn;
        S[m][st * 16 + nn] = (jp <= qp) ? val * 0.125f : -3.0e38f;
      });
    }
    __syncthreads();
    if (lane < 16) {
      float mx = run_max;
      for (int j2 = 0; j2 < 32; ++j2) mx = fmaxf(mx, S[lane][j2]);
      const float corr = __expf(run_max - mx);
      float ts = 0.f;
      for (int j2 = 0; j2 < 32; ++j2) {
        const float e = __expf(S[lane][j2] - mx);
        P[lane][j2] = to_bf16(e);
        ts += e;
      }
      run_sum = run_sum * corr + ts;
      run_max = mx;
      f_corr[lane] = corr;
    }
    __syncthreads();
    v16bf pa = load_a_frag([&](int m, int kk) { return bf16_to_f32(P[m][kk]); });
#pragma unroll
    for (int ntv = 0; ntv < 4; ++ntv) {
#pragma unroll
      for (int r = 0; r < 8; ++r) oacc[ntv][r] *= f_corr[r + 8 * (lane >> 4)];
      v16bf vb = load_b_frag([&](int kk, int nn) {
        return bf16_to_f32(vbuf[(size_t)kk * HPd + ntv * 16 + nn]);
      });
      oacc[ntv] = wmma_bf16(pa, vb, oacc[ntv]);
    }
    __syncthreads();
  }
  if (lane < 16) f_inv[lane] = 1.f / run_sum;
  __syncthreads();
#pragma unroll
  for (int ntv = 0; ntv < 4; ++ntv)
    store_c(oacc[ntv], [&](int m, int nn, float val) {
      const int t = qt * 16 + m;
      ctx[((size_t)(b * Tt + t)) * CTXW + n * HPd + ntv * 16 + nn] =
          val * f_inv[m];
    });
}

// ---- kernel 8: fused output GEMM: out = mctx*mwo + dctx*dwo -----------------
// grid (256 row tiles, 64 col tiles); block 32
__global__ void k_out(const float* __restrict__ mctx,
                      const float* __restrict__ mwo,
                      const float* __restrict__ dctx,
                      const float* __restrict__ dwo,
                      float* __restrict__ out) {
  const int rt = blockIdx.x, ct = blockIdx.y;
  const int lane = threadIdx.x & 31;
  v8f acc = {};
  const float* ar = mctx + (size_t)(rt * 16 + (lane & 15)) * CTXW;
  for (int kc = 0; kc < CTXW; kc += 32) {
    __builtin_prefetch(ar + kc + 128, 0, 1);
    v16bf a = load_a_frag([&](int, int kk) { return ar[kc + kk]; });
    v16bf w = load_b_frag([&](int kk, int nn) {
      return mwo[(size_t)(kc + kk) * Hh + ct * 16 + nn];
    });
    acc = wmma_bf16(a, w, acc);
  }
  const float* dr = dctx + (size_t)(rt * 16 + (lane & 15)) * CTXW;
  for (int kc = 0; kc < CTXW; kc += 32) {
    __builtin_prefetch(dr + kc + 128, 0, 1);
    v16bf a = load_a_frag([&](int, int kk) { return dr[kc + kk]; });
    v16bf w = load_b_frag([&](int kk, int nn) {
      return dwo[(size_t)(kc + kk) * Hh + ct * 16 + nn];
    });
    acc = wmma_bf16(a, w, acc);
  }
  store_c(acc, [&](int m, int nn, float val) {
    out[(size_t)(rt * 16 + m) * Hh + ct * 16 + nn] = val;
  });
}

// ---- launcher ---------------------------------------------------------------
extern "C" void kernel_launch(void* const* d_in, const int* in_sizes, int n_in,
                              void* d_out, int out_size, void* d_ws,
                              size_t ws_size, hipStream_t stream) {
  const float* X    = (const float*)d_in[0];
  const float* mrw  = (const float*)d_in[1];
  const float* mwqkv= (const float*)d_in[2];
  const float* mwo  = (const float*)d_in[3];
  const float* dwqkv= (const float*)d_in[4];
  const float* dwo  = (const float*)d_in[5];
  float* out = (float*)d_out;

  char* ws = (char*)d_ws;
  size_t off = 0;
  auto carve = [&](size_t bytes) {
    size_t o = off;
    off = (off + bytes + 255) & ~(size_t)255;
    return o;
  };
  int*    idx   = (int*)   (ws + carve((size_t)Bb * NHM * KSEL * 4));
  float*  gates = (float*) (ws + carve((size_t)Bb * NHM * KSEL * 4));
  __bf16* mQ    = (__bf16*)(ws + carve((size_t)Bb * NHM * KSEL * HPd * 2));
  __bf16* mK    = (__bf16*)(ws + carve((size_t)Bb * NHM * KSEL * HPd * 2));
  __bf16* mV    = (__bf16*)(ws + carve((size_t)Bb * NHM * KSEL * HPd * 2));
  float*  mctx  = (float*) (ws + carve((size_t)Bb * Tt * CTXW * 4));
  __bf16* dQ    = (__bf16*)(ws + carve((size_t)Bb * NHD * Tt * HPd * 2));
  __bf16* dK    = (__bf16*)(ws + carve((size_t)Bb * NHD * Tt * HPd * 2));
  __bf16* dV    = (__bf16*)(ws + carve((size_t)Bb * NHD * Tt * HPd * 2));
  float*  dctx  = (float*) (ws + carve((size_t)Bb * Tt * CTXW * 4));
  (void)ws_size; (void)in_sizes; (void)n_in; (void)out_size;

  // MoSA branch
  k_router_topk<<<Bb * NHM, 256, 0, stream>>>(X, mrw, idx, gates);
  k_zero<<<2048, 256, 0, stream>>>(mctx, Bb * Tt * CTXW);
  k_mosa_qkv<<<dim3(16, 12, Bb * NHM), 32, 0, stream>>>(X, mwqkv, idx, mQ, mK, mV);
  k_mosa_rope<<<(Bb * NHM * KSEL * RHALF + 255) / 256, 256, 0, stream>>>(mQ, mK, idx);
  k_mosa_attn<<<Bb * NHM * 16, 32, 0, stream>>>(mQ, mK, mV, idx, gates, mctx);

  // Dense branch
  k_dense_qkv<<<dim3(256, 96), 32, 0, stream>>>(X, dwqkv, dQ, dK, dV);
  k_dense_rope<<<(Bb * NHD * Tt * RHALF + 255) / 256, 256, 0, stream>>>(dQ, dK);
  k_dense_attn<<<dim3(Tt / 16, Bb * NHD), 32, 0, stream>>>(dQ, dK, dV, dctx);

  // Fused output projections + sum
  k_out<<<dim3((Bb * Tt) / 16, Hh / 16), 32, 0, stream>>>(mctx, mwo, dctx, dwo, out);
}
